// RNNBase_80281528697416
// MI455X (gfx1250) — matrix operand order repaired and checked by reading, not compiled, not run on hardware
//
#include <hip/hip_runtime.h>
#include <stdint.h>

#define B_ 32
#define T_ 512
#define D_ 512
#define H_ 512
#define G_ 2048   // 4*H

typedef __attribute__((ext_vector_type(16))) __bf16 v16bf;
typedef __attribute__((ext_vector_type(8)))  float  v8f;

union Frag32B { uint4 q[2]; v16bf v; };

__device__ __forceinline__ unsigned short f2bf(float f) {
  unsigned u = __float_as_uint(f);
  u += 0x7FFFu + ((u >> 16) & 1u);           // round-to-nearest-even
  return (unsigned short)(u >> 16);
}
__device__ __forceinline__ float sigmoidf_(float x) { return 1.0f / (1.0f + __expf(-x)); }
__device__ __forceinline__ float tanhf_(float x) {
  float e = __expf(-2.0f * x);
  return (1.0f - e) / (1.0f + e);
}

// ---------------- conversion kernels ----------------
__global__ void k_f32_to_bf16(const float* __restrict__ src, unsigned short* __restrict__ dst, int n) {
  for (int i = blockIdx.x * blockDim.x + threadIdx.x; i < n; i += gridDim.x * blockDim.x)
    dst[i] = f2bf(src[i]);
}

__global__ void k_bias_combine(const float* __restrict__ a, const float* __restrict__ b,
                               float* __restrict__ dst, int n) {
  for (int i = blockIdx.x * blockDim.x + threadIdx.x; i < n; i += gridDim.x * blockDim.x)
    dst[i] = a[i] + b[i];
}

// zero c-state, zero h(t=-1) slot, reset barrier counters
__global__ void k_init_rec(float* __restrict__ cstate, unsigned short* __restrict__ hseq,
                           unsigned* __restrict__ cnt, unsigned* __restrict__ gen) {
  int i = blockIdx.x * blockDim.x + threadIdx.x;
  if (i == 0) { *cnt = 0u; *gen = 0u; }
  if (i < B_ * H_) {
    cstate[i] = 0.0f;
    int b = i >> 9, j = i & (H_ - 1);
    hseq[((size_t)b * (T_ + 1)) * H_ + j] = 0;   // h_{-1} = 0
  }
}

// ---------------- big WMMA GEMM: Cout[m][n] = A_bf16[m,:] . W_bf16[n,:] + bias[n] ----------------
// A: bf16 rows of length K (row-major). remap!=0 => A rows live in hseq layout [B, T+1, H], slot t+1.
__global__ void __launch_bounds__(256)
k_wmma_gemm(const unsigned short* __restrict__ A,
            const unsigned short* __restrict__ W,     // [G_][K] row-major
            const float* __restrict__ bias,           // [G_]
            float* __restrict__ Cout,                 // [M][G_]
            int M, int K, int remap) {
  const int lane = threadIdx.x & 31;
  const int l15 = lane & 15;
  const int hs = lane >> 4;                       // 0 or 1
  const int wid = blockIdx.x * (blockDim.x >> 5) + (threadIdx.x >> 5);
  const int nwaves = gridDim.x * (blockDim.x >> 5);
  const int NG = G_ / 64;                         // groups of 4 n-tiles
  const int tasks = (M / 16) * NG;
  for (int w = wid; w < tasks; w += nwaves) {
    int mt = w / NG;
    int ng = w - mt * NG;
    int m0 = mt * 16;
    int n0 = ng * 64;
    int arow0;
    if (remap) { int b = m0 / T_; int t0 = m0 - b * T_; arow0 = b * (T_ + 1) + t0 + 1; }
    else       { arow0 = m0; }
    const unsigned short* arow = A + (size_t)(arow0 + l15) * K;
    v8f acc[4] = {};
    for (int kk = 0; kk < K; kk += 32) {
      Frag32B fa;
      const unsigned short* pa = arow + kk + hs * 8;
      fa.q[0] = *(const uint4*)(pa);
      fa.q[1] = *(const uint4*)(pa + 16);
#pragma unroll
      for (int s = 0; s < 4; ++s) {
        Frag32B fb;
        const unsigned short* pb = W + (size_t)(n0 + s * 16 + l15) * K + kk + hs * 8;
        fb.q[0] = *(const uint4*)(pb);
        fb.q[1] = *(const uint4*)(pb + 16);
        acc[s] = __builtin_amdgcn_wmma_f32_16x16x32_bf16(
            false, fa.v, false, fb.v, (short)0, acc[s], false, false);
      }
    }
#pragma unroll
    for (int s = 0; s < 4; ++s) {
      int n = n0 + s * 16 + l15;
      float bv = bias[n];
#pragma unroll
      for (int r = 0; r < 8; ++r) {
        int m = m0 + r + 8 * hs;
        Cout[(size_t)m * G_ + n] = acc[s][r] + bv;
      }
    }
  }
}

// ---------------- persistent LSTM recurrence ----------------
// 16 blocks x 128 threads (4 waves). wid = 0..63; jt = wid>>1 (column tile of H),
// mt = wid&1 (batch half). Each wave computes ALL FOUR gate tiles for its (mt, jt)
// so the cell update is register-local; only ONE grid barrier per timestep.
// W_hh rows for the block's two jt values are staged in 128 KB of LDS once.
#define RNB 16
#define RNT 128
#define LDS_USHORTS (2 * 64 * H_)   // 2 regions * 64 rows * 512 = 65536 ushorts = 128 KB

__device__ __forceinline__ void gridbar(unsigned* cnt, unsigned* gen, unsigned target) {
  __syncthreads();
  if (threadIdx.x == 0) {
    __threadfence();
    if (atomicAdd(cnt, 1u) == (unsigned)(RNB - 1)) {
      *((volatile unsigned*)cnt) = 0u;
      __threadfence();
      atomicAdd(gen, 1u);
    } else {
      while (*((volatile unsigned*)gen) < target) __builtin_amdgcn_s_sleep(2);
    }
    __threadfence();
  }
  __syncthreads();
}

__global__ void __launch_bounds__(RNT)
k_lstm_recur(const float* __restrict__ xw,           // [B*T][G_]  (bias already folded in)
             unsigned short* __restrict__ hseq,      // bf16 [B][T+1][H]
             const unsigned short* __restrict__ whh, // bf16 [G_][H]
             float* __restrict__ cstate,             // [B*H]
             float* __restrict__ out,                // [B][T][H] fp32 or nullptr
             unsigned* __restrict__ cnt, unsigned* __restrict__ gen) {
  extern __shared__ unsigned short smem[];           // LDS_USHORTS
  const int lane = threadIdx.x & 31;
  const int l15 = lane & 15;
  const int hs = lane >> 4;
  const int wib = threadIdx.x >> 5;                  // 0..3
  const int wid = blockIdx.x * (RNT >> 5) + wib;     // 0..63
  const int jt = wid >> 1;                           // 0..31  (16-col group of H)
  const int mt = wid & 1;                            // batch half
  const int region = wib >> 1;                       // which LDS region this wave uses

  // ---- stage W_hh rows into LDS (cooperative, once) ----
  // region r holds 64 rows: n = g*512 + (blockIdx.x*2 + r)*16 + rr, local row lr = g*16+rr.
  {
    uint4* s4 = (uint4*)smem;
    const int jt0 = blockIdx.x * 2;
    for (int i = threadIdx.x; i < 2 * 64 * (H_ / 8); i += RNT) {  // 8192 uint4
      int row = i >> 6;                 // 0..127 (64 uint4 per row)
      int q   = i & 63;
      int reg = row >> 6;               // 0 or 1
      int lr  = row & 63;
      int g   = lr >> 4;
      int rr  = lr & 15;
      int n   = g * 512 + (jt0 + reg) * 16 + rr;
      s4[i] = ((const uint4*)(whh + (size_t)n * H_))[q];
    }
    __syncthreads();
  }

  const unsigned short* sw = smem + region * (64 * H_);
  const int j = jt * 16 + l15;                       // this lane's hidden column

  for (int t = 0; t < T_; ++t) {
    // init accumulators from precomputed xW (+bias)
    v8f acc[4];
#pragma unroll
    for (int g = 0; g < 4; ++g) {
      int n = g * 512 + j;
#pragma unroll
      for (int r = 0; r < 8; ++r) {
        int b = mt * 16 + r + 8 * hs;
        acc[g][r] = xw[((size_t)b * T_ + t) * G_ + n];
      }
    }
    // accumulate h_{t-1} @ W_hh^T : A from global (bf16 hseq), B from LDS
    const unsigned short* ha = hseq + ((size_t)(mt * 16 + l15) * (T_ + 1) + t) * H_;
    for (int kk = 0; kk < H_; kk += 32) {
      Frag32B fa;
      const unsigned short* pa = ha + kk + hs * 8;
      fa.q[0] = *(const uint4*)(pa);
      fa.q[1] = *(const uint4*)(pa + 16);
#pragma unroll
      for (int g = 0; g < 4; ++g) {
        Frag32B fb;
        const unsigned short* pb = sw + (size_t)(g * 16 + l15) * H_ + kk + hs * 8;
        fb.q[0] = *(const uint4*)(pb);
        fb.q[1] = *(const uint4*)(pb + 16);
        acc[g] = __builtin_amdgcn_wmma_f32_16x16x32_bf16(
            false, fa.v, false, fb.v, (short)0, acc[g], false, false);
      }
    }
    // register-local cell update: lane holds i,f,g,o for its own (b, j) elements
#pragma unroll
    for (int r = 0; r < 8; ++r) {
      int b = mt * 16 + r + 8 * hs;
      size_t e = (size_t)b * H_ + j;
      float gi = sigmoidf_(acc[0][r]);
      float gf = sigmoidf_(acc[1][r]);
      float gg = tanhf_(acc[2][r]);
      float go = sigmoidf_(acc[3][r]);
      float c  = gf * cstate[e] + gi * gg;
      cstate[e] = c;
      float h = go * tanhf_(c);
      hseq[((size_t)b * (T_ + 1) + t + 1) * H_ + j] = f2bf(h);
      if (out) out[((size_t)b * T_ + t) * H_ + j] = h;
    }
    gridbar(cnt, gen, (unsigned)t + 1u);   // h(t) visible before anyone reads it at t+1
  }
}

// ---------------- launcher ----------------
extern "C" void kernel_launch(void* const* d_in, const int* in_sizes, int n_in,
                              void* d_out, int out_size, void* d_ws, size_t ws_size,
                              hipStream_t stream) {
  const float* x_f    = (const float*)d_in[0];   // [B, T, D]
  const float* Wih0_f = (const float*)d_in[1];   // [4H, D]
  const float* Whh0_f = (const float*)d_in[2];   // [4H, H]
  const float* bih0   = (const float*)d_in[3];
  const float* bhh0   = (const float*)d_in[4];
  const float* Wih1_f = (const float*)d_in[5];   // [4H, H]
  const float* Whh1_f = (const float*)d_in[6];
  const float* bih1   = (const float*)d_in[7];
  const float* bhh1   = (const float*)d_in[8];

  char* ws = (char*)d_ws;
  size_t o = 0;
  auto take = [&](size_t bytes) { size_t r = o; o = (o + bytes + 255) & ~(size_t)255; return r; };

  unsigned short* wih0_bf = (unsigned short*)(ws + take((size_t)G_ * D_ * 2));
  unsigned short* whh0_bf = (unsigned short*)(ws + take((size_t)G_ * H_ * 2));
  unsigned short* wih1_bf = (unsigned short*)(ws + take((size_t)G_ * H_ * 2));
  unsigned short* whh1_bf = (unsigned short*)(ws + take((size_t)G_ * H_ * 2));
  float*          bias0   = (float*)(ws + take((size_t)G_ * 4));
  float*          bias1   = (float*)(ws + take((size_t)G_ * 4));
  unsigned short* x_bf    = (unsigned short*)(ws + take((size_t)B_ * T_ * D_ * 2));
  unsigned short* hseq    = (unsigned short*)(ws + take((size_t)B_ * (T_ + 1) * H_ * 2));
  float*          xw      = (float*)(ws + take((size_t)B_ * T_ * G_ * 4));
  float*          cstate  = (float*)(ws + take((size_t)B_ * H_ * 4));
  unsigned*       cnt     = (unsigned*)(ws + take(256));
  unsigned*       gen     = cnt + 1;

  // convert weights / biases / input
  k_f32_to_bf16<<<1024, 256, 0, stream>>>(Wih0_f, wih0_bf, G_ * D_);
  k_f32_to_bf16<<<1024, 256, 0, stream>>>(Whh0_f, whh0_bf, G_ * H_);
  k_f32_to_bf16<<<1024, 256, 0, stream>>>(Wih1_f, wih1_bf, G_ * H_);
  k_f32_to_bf16<<<1024, 256, 0, stream>>>(Whh1_f, whh1_bf, G_ * H_);
  k_bias_combine<<<8, 256, 0, stream>>>(bih0, bhh0, bias0, G_);
  k_bias_combine<<<8, 256, 0, stream>>>(bih1, bhh1, bias1, G_);
  k_f32_to_bf16<<<4096, 256, 0, stream>>>(x_f, x_bf, B_ * T_ * D_);

  const size_t ldsBytes = (size_t)LDS_USHORTS * 2;   // 128 KB

  // layer 0: input projection (one big WMMA GEMM), then recurrence
  k_wmma_gemm<<<4096, 256, 0, stream>>>(x_bf, wih0_bf, bias0, xw, B_ * T_, D_, 0);
  k_init_rec<<<64, 256, 0, stream>>>(cstate, hseq, cnt, gen);
  k_lstm_recur<<<RNB, RNT, ldsBytes, stream>>>(xw, hseq, whh0_bf, cstate, nullptr, cnt, gen);

  // layer 1: input projection from h0_seq (bf16, remapped rows), then recurrence -> d_out
  k_wmma_gemm<<<4096, 256, 0, stream>>>(hseq, wih1_bf, bias1, xw, B_ * T_, H_, 1);
  k_init_rec<<<64, 256, 0, stream>>>(cstate, hseq, cnt, gen);
  k_lstm_recur<<<RNB, RNT, ldsBytes, stream>>>(xw, hseq, whh1_bf, cstate, (float*)d_out, cnt, gen);

  (void)in_sizes; (void)n_in; (void)out_size; (void)ws_size;
}
